// HeteroRGCN_45054206935552
// MI455X (gfx1250) — compile-verified, matrix operand order checked by reading, not run on hardware
//
#include <hip/hip_runtime.h>

typedef __attribute__((ext_vector_type(16))) _Float16 v16h;
typedef __attribute__((ext_vector_type(8)))  float    v8f;

#define N_TX 1000000
#define N_W  250000
#define NE   2000000
#define ROWS 128      // rows per block (8 waves x 16 rows)
#define LSTR 72       // LDS row stride in elements (144B rows: 16B aligned, bank-spread)

union FragU { uint4 u[2]; v16h h; };

// Load a 16-element f16 WMMA fragment from an LDS row laid out [idx][k] with
// stride LSTR. Per ISA 16-bit A/B layout: elements 0..7 = K kb..kb+7,
// elements 8..15 = K kb+16..kb+23 (kb already includes the lane>=16 +8 shift).
__device__ __forceinline__ v16h load_frag(const _Float16* p) {
    FragU r;
    r.u[0] = *(const uint4*)(p);
    r.u[1] = *(const uint4*)(p + 16);
    return r.h;
}

// out[r, 0:NOUT) = act( (A1[r,:]/max(cnt[r],1)) @ W1 + (DUAL ? A2[r,:] @ W2 : 0) + bias )
// A1 row stride == K; A2 row stride == 64 (only used when DUAL, K==64).
// W1 row-major [K][NOUT]; W2 row-major [64][64]. Output row stride == NOUT.
// All buffers < 2^31 elements -> 32-bit offset arithmetic throughout.
template<int K, int NOUT, bool DUAL, bool RELU>
__global__ __launch_bounds__(256) void gemm_h(
    const float* __restrict__ A1, const float* __restrict__ cnt,
    const float* __restrict__ W1, const float* __restrict__ bias,
    const float* __restrict__ A2, const float* __restrict__ W2,
    float* __restrict__ out, int M)
{
    constexpr int NTILES = (NOUT + 15) / 16;

    __shared__ alignas(16) _Float16 As1[ROWS * LSTR];
    __shared__ alignas(16) _Float16 Wt1[64 * LSTR];
    __shared__ alignas(16) _Float16 As2[DUAL ? ROWS * LSTR : 8];
    __shared__ alignas(16) _Float16 Wt2[DUAL ? 64 * LSTR : 8];

    const int tid = threadIdx.x;
    const int r0  = blockIdx.x * ROWS;

    // Stage weights transposed into LDS: Wt[n][k] = W[k][n] (f32 -> f16)
    for (int i = tid; i < K * NOUT; i += 256) {
        const int k = i / NOUT, n = i - k * NOUT;
        Wt1[n * LSTR + k] = (_Float16)W1[i];
    }
    if constexpr (DUAL) {
        for (int i = tid; i < 64 * 64; i += 256) {
            const int k = i >> 6, n = i & 63;
            Wt2[n * LSTR + k] = (_Float16)W2[i];
        }
    }

    // Stage activation rows into LDS as f16 (mean-divide folded in when DUAL)
    for (int i = tid; i < ROWS * K / 4; i += 256) {
        const int e = i * 4;
        const int m = e / K, k = e - m * K;
        const int gm = r0 + m;
        float4 v = make_float4(0.f, 0.f, 0.f, 0.f);
        float s = 1.0f;
        if (gm < M) {
            v = *(const float4*)(A1 + gm * K + k);
            if constexpr (DUAL) s = 1.0f / fmaxf(cnt[gm], 1.0f);
        }
        _Float16* p = &As1[m * LSTR + k];
        p[0] = (_Float16)(v.x * s); p[1] = (_Float16)(v.y * s);
        p[2] = (_Float16)(v.z * s); p[3] = (_Float16)(v.w * s);
        if constexpr (DUAL) {
            float4 u = make_float4(0.f, 0.f, 0.f, 0.f);
            if (gm < M) u = *(const float4*)(A2 + gm * 64 + k);
            _Float16* q = &As2[m * LSTR + k];
            q[0] = (_Float16)u.x; q[1] = (_Float16)u.y;
            q[2] = (_Float16)u.z; q[3] = (_Float16)u.w;
        }
    }
    __syncthreads();

    const int lane  = tid & 31;
    const int wave  = tid >> 5;
    const int mrow  = wave * 16 + (lane & 15);   // A row this lane supplies
    const int khalf = (lane >> 4) << 3;          // lanes 16..31 hold the K+8 group

    v8f acc[NTILES] = {};

    #pragma unroll
    for (int ks = 0; ks < K / 32; ks++) {
        const int kb = ks * 32 + khalf;
        const v16h a1 = load_frag(&As1[mrow * LSTR + kb]);
        v16h a2 = a1;
        if constexpr (DUAL) a2 = load_frag(&As2[mrow * LSTR + kb]);
        #pragma unroll
        for (int nt = 0; nt < NTILES; nt++) {
            const int ncol = nt * 16 + (lane & 15);
            const v16h b1 = load_frag(&Wt1[ncol * LSTR + kb]);
            acc[nt] = __builtin_amdgcn_wmma_f32_16x16x32_f16(
                false, a1, false, b1, (short)0, acc[nt], false, false);
            if constexpr (DUAL) {
                const v16h b2 = load_frag(&Wt2[ncol * LSTR + kb]);
                acc[nt] = __builtin_amdgcn_wmma_f32_16x16x32_f16(
                    false, a2, false, b2, (short)0, acc[nt], false, false);
            }
        }
    }

    // C/D layout: VGPR v -> row (v + 8*(lane>=16)), col = lane&15 (per N tile).
    // Fast path (all but the last block): unpredicated stores at compile-time
    // immediate offsets v*NOUT*4 from a per-lane base pointer (32-bit offsets).
    const int  base_row = r0 + wave * 16 + ((lane >> 4) << 3);
    const bool full     = (r0 + ROWS) <= M;

    #pragma unroll
    for (int nt = 0; nt < NTILES; nt++) {
        const int n = nt * 16 + (lane & 15);
        if (NOUT < 16 && n >= NOUT) continue;   // only trims lanes when NOUT==2
        const float b = bias[n];
        float* __restrict__ p = out + base_row * NOUT + n;
        if (full) {
            #pragma unroll
            for (int v = 0; v < 8; v++) {
                float x = acc[nt][v] + b;
                if (RELU) x = fmaxf(x, 0.f);
                p[v * NOUT] = x;
            }
        } else {
            #pragma unroll
            for (int v = 0; v < 8; v++) {
                if (base_row + v < M) {
                    float x = acc[nt][v] + b;
                    if (RELU) x = fmaxf(x, 0.f);
                    p[v * NOUT] = x;
                }
            }
        }
    }
}

// Edge scatter: agg[dst] += x[src] (64 f32 cols), cnt[dst] += 1.
// One thread per (edge, 4-column chunk) -> float4 gather + 4 f32 atomics.
__global__ __launch_bounds__(256) void scatter_mean(
    const float* __restrict__ x, const int* __restrict__ src,
    const int* __restrict__ dst, float* __restrict__ agg,
    float* __restrict__ cnt, int E)
{
    const int g = blockIdx.x * 256 + threadIdx.x;
    const int total = E * 16;
    if (g >= total) return;
    const int e  = g >> 4;
    const int c4 = (g & 15) << 2;
    const int s = src[e], d = dst[e];
    const float4 v = *(const float4*)(x + s * 64 + c4);
    float* o = agg + d * 64 + c4;
    atomicAdd(o + 0, v.x);
    atomicAdd(o + 1, v.y);
    atomicAdd(o + 2, v.z);
    atomicAdd(o + 3, v.w);
    if ((g & 15) == 0) atomicAdd(cnt + d, 1.0f);
}

extern "C" void kernel_launch(void* const* d_in, const int* in_sizes, int n_in,
                              void* d_out, int out_size, void* d_ws, size_t ws_size,
                              hipStream_t stream)
{
    (void)in_sizes; (void)n_in; (void)out_size; (void)ws_size;

    const float* x_tx   = (const float*)d_in[0];
    const float* x_w    = (const float*)d_in[1];
    const int*   src_tw = (const int*)d_in[2];
    const int*   dst_tw = (const int*)d_in[3];
    const int*   src_wt = (const int*)d_in[4];
    const int*   dst_wt = (const int*)d_in[5];
    const float* Win_tx = (const float*)d_in[6];  const float* bin_tx = (const float*)d_in[7];
    const float* Win_w  = (const float*)d_in[8];  const float* bin_w  = (const float*)d_in[9];
    const float* Wl1_tw = (const float*)d_in[10]; const float* bl1_tw = (const float*)d_in[11]; const float* Wr1_tw = (const float*)d_in[12];
    const float* Wl1_wt = (const float*)d_in[13]; const float* bl1_wt = (const float*)d_in[14]; const float* Wr1_wt = (const float*)d_in[15];
    const float* Wl2_tw = (const float*)d_in[16]; const float* bl2_tw = (const float*)d_in[17]; const float* Wr2_tw = (const float*)d_in[18];
    const float* Wl2_wt = (const float*)d_in[19]; const float* bl2_wt = (const float*)d_in[20]; const float* Wr2_wt = (const float*)d_in[21];
    const float* Wh1 = (const float*)d_in[22]; const float* bh1 = (const float*)d_in[23];
    const float* Wh2 = (const float*)d_in[24]; const float* bh2 = (const float*)d_in[25];

    // d_out = logits[N_TX,2] ++ t2[N_TX,64] ++ w2[N_W,64]
    float* out        = (float*)d_out;
    float* out_logits = out;
    float* out_t2     = out + 2L * N_TX;
    float* out_w2     = out + 2L * N_TX + 64L * N_TX;

    // workspace layout (floats)
    float* ws     = (float*)d_ws;
    float* h_tx   = ws;                   // 64e6   (reused as head hidden)
    float* h_w    = ws + 64000000L;       // 16e6
    float* t1     = ws + 80000000L;       // 64e6
    float* w1     = ws + 144000000L;      // 16e6
    float* agg_tx = ws + 160000000L;      // 64e6
    float* cnt_tx = ws + 224000000L;      // 1e6   (contiguous after agg_tx)
    float* agg_w  = ws + 225000000L;      // 16e6
    float* cnt_w  = ws + 241000000L;      // 0.25e6 (contiguous after agg_w)

    const dim3 blk(256);
    const int gbTX = (N_TX + ROWS - 1) / ROWS;
    const int gbW  = (N_W  + ROWS - 1) / ROWS;
    const int gbE  = (NE * 16 + 255) / 256;

    // per-type input projections
    gemm_h<64, 64, false, false><<<gbTX, blk, 0, stream>>>(
        x_tx, nullptr, Win_tx, bin_tx, nullptr, nullptr, h_tx, N_TX);
    gemm_h<32, 64, false, false><<<gbW, blk, 0, stream>>>(
        x_w, nullptr, Win_w, bin_w, nullptr, nullptr, h_w, N_W);

    // conv1: w1 = relu(SAGE(h_tx -> W via tw)), t1 = relu(SAGE(h_w -> TX via wt))
    hipMemsetAsync(agg_w, 0, (16000000L + 250000L) * sizeof(float), stream);
    scatter_mean<<<gbE, blk, 0, stream>>>(h_tx, src_tw, dst_tw, agg_w, cnt_w, NE);
    gemm_h<64, 64, true, true><<<gbW, blk, 0, stream>>>(
        agg_w, cnt_w, Wl1_tw, bl1_tw, h_w, Wr1_tw, w1, N_W);

    hipMemsetAsync(agg_tx, 0, (64000000L + 1000000L) * sizeof(float), stream);
    scatter_mean<<<gbE, blk, 0, stream>>>(h_w, src_wt, dst_wt, agg_tx, cnt_tx, NE);
    gemm_h<64, 64, true, true><<<gbTX, blk, 0, stream>>>(
        agg_tx, cnt_tx, Wl1_wt, bl1_wt, h_tx, Wr1_wt, t1, N_TX);

    // conv2: w2, t2 (written straight to d_out)
    hipMemsetAsync(agg_w, 0, (16000000L + 250000L) * sizeof(float), stream);
    scatter_mean<<<gbE, blk, 0, stream>>>(t1, src_tw, dst_tw, agg_w, cnt_w, NE);
    gemm_h<64, 64, true, true><<<gbW, blk, 0, stream>>>(
        agg_w, cnt_w, Wl2_tw, bl2_tw, w1, Wr2_tw, out_w2, N_W);

    hipMemsetAsync(agg_tx, 0, (64000000L + 1000000L) * sizeof(float), stream);
    scatter_mean<<<gbE, blk, 0, stream>>>(w1, src_wt, dst_wt, agg_tx, cnt_tx, NE);
    gemm_h<64, 64, true, true><<<gbTX, blk, 0, stream>>>(
        agg_tx, cnt_tx, Wl2_wt, bl2_wt, t1, Wr2_wt, out_t2, N_TX);

    // tx head: hidden = relu(t2 @ Wh1 + bh1); logits = hidden @ Wh2 + bh2
    float* hidden = h_tx;  // h_tx is dead past conv1-wt; reuse
    gemm_h<64, 64, false, true><<<gbTX, blk, 0, stream>>>(
        out_t2, nullptr, Wh1, bh1, nullptr, nullptr, hidden, N_TX);
    gemm_h<64, 2, false, false><<<gbTX, blk, 0, stream>>>(
        hidden, nullptr, Wh2, bh2, nullptr, nullptr, out_logits, N_TX);
}
